// MultiHeadAttention_58626303590702
// MI455X (gfx1250) — matrix operand order repaired and checked by reading, not compiled, not run on hardware
//
#include <hip/hip_runtime.h>

// MI455X (gfx1250) fused multi-head attention, bf16 WMMA + TDM async tensor loads.
// B=4, S=2048, D=1024, H=16, HD=64.

#define BD   4
#define SD   2048
#define DDIM 1024
#define HH   16
#define HDIM 64

typedef __attribute__((ext_vector_type(16))) __bf16 bf16x16;
typedef __attribute__((ext_vector_type(8)))  __bf16 bf16x8;
typedef __attribute__((ext_vector_type(8)))  float  f32x8;
typedef __attribute__((ext_vector_type(4)))  unsigned int   uint4v;
typedef __attribute__((ext_vector_type(8)))  int            int8v;
typedef __attribute__((ext_vector_type(4)))  int            int4v;

__device__ __forceinline__ f32x8 zero8() {
  f32x8 z;
  for (int i = 0; i < 8; ++i) z[i] = 0.0f;
  return z;
}

// f32 -> bf16 round-to-nearest-even
__device__ __forceinline__ unsigned short f2bf(float f) {
  unsigned u = __builtin_bit_cast(unsigned, f);
  u += 0x7FFFu + ((u >> 16) & 1u);
  return (unsigned short)(u >> 16);
}

__device__ __forceinline__ f32x8 wmma_bf16(bf16x16 a, bf16x16 b, f32x8 c) {
  return __builtin_amdgcn_wmma_f32_16x16x32_bf16(
      false, a, false, b, (short)0, c, false, false);
}

// A-fragment (16x32, MxK). Row-major source [row][k], stride ld (elements).
__device__ __forceinline__ bf16x16 load_afrag(const unsigned short* base, long ld,
                                              int row0, int k0) {
  const int lane = threadIdx.x & 31;
  const int m = lane & 15, h2 = lane >> 4;
  const unsigned short* p = base + (long)(row0 + m) * ld + k0 + h2 * 8;
  bf16x8 lo = *reinterpret_cast<const bf16x8*>(p);
  bf16x8 hi = *reinterpret_cast<const bf16x8*>(p + 16);
  bf16x16 a;
  for (int j = 0; j < 8; ++j) { a[j] = lo[j]; a[j + 8] = hi[j]; }
  return a;
}

// B-fragment (32x16, KxN) from source laid out as [n][k] rows (stride ld).
__device__ __forceinline__ bf16x16 load_bfrag(const unsigned short* base, long ld,
                                              int row0, int k0) {
  const int lane = threadIdx.x & 31;
  const int n = lane & 15, h2 = lane >> 4;
  const unsigned short* p = base + (long)(row0 + n) * ld + k0 + h2 * 8;
  bf16x8 lo = *reinterpret_cast<const bf16x8*>(p);
  bf16x8 hi = *reinterpret_cast<const bf16x8*>(p + 16);
  bf16x16 b;
  for (int j = 0; j < 8; ++j) { b[j] = lo[j]; b[j + 8] = hi[j]; }
  return b;
}

// ---------------------------------------------------------------------------
// Tensor Data Mover: async 2D tile load (bf16 elements) global -> LDS.
// D# built per CDNA5 ISA. Issued by one wave.
// pad_bits: OR of pad_enable[20] | pad_interval[24:22] | pad_amount[31:25].
// ---------------------------------------------------------------------------
__device__ __forceinline__ void tdm_load_2d(const void* lds_ptr, const void* gptr,
                                            unsigned tensor_d0, unsigned tensor_d1,
                                            unsigned stride0,
                                            unsigned tile_d0, unsigned tile_d1,
                                            unsigned pad_bits = 0u) {
  unsigned lds_off = (unsigned)(unsigned long long)lds_ptr;  // low 32b of flat ptr
  unsigned long long ga = (unsigned long long)gptr;
  uint4v g0;
  g0[0] = 1u;                                               // count=1, user D#
  g0[1] = lds_off;                                          // lds_addr
  g0[2] = (unsigned)(ga & 0xFFFFFFFFu);                     // global_addr[31:0]
  g0[3] = (unsigned)((ga >> 32) & 0x01FFFFFFu) | (2u << 30);  // addr[56:32]|type=2
  int8v g1;
  g1[0] = (int)((1u << 16) | pad_bits);                     // data_size=1 (2B)
  g1[1] = (int)((tensor_d0 & 0xFFFFu) << 16);               // tensor_dim0[15:0]
  g1[2] = (int)((tensor_d0 >> 16) | ((tensor_d1 & 0xFFFFu) << 16));
  g1[3] = (int)((tensor_d1 >> 16) | (tile_d0 << 16));       // tile_dim0
  g1[4] = (int)(tile_d1 & 0xFFFFu);                         // tile_dim1 (dim2=0)
  g1[5] = (int)stride0;                                     // dim0_stride[31:0]
  g1[6] = 0;
  g1[7] = 0;
  int4v z4;
  z4[0] = 0; z4[1] = 0; z4[2] = 0; z4[3] = 0;
#if __clang_major__ >= 23
  int8v z8;
  for (int i = 0; i < 8; ++i) z8[i] = 0;
  __builtin_amdgcn_tensor_load_to_lds(g0, g1, z4, z4, z8, 0);
#else
  __builtin_amdgcn_tensor_load_to_lds(g0, g1, z4, z4, 0);
#endif
}
// pad: after every 16 DWORDs (code 3) insert 4 DWORDs (code 3) -> 40-ushort rows
#define TDM_PAD_40 ((1u << 20) | (3u << 22) | (3u << 25))

// ---------------------------------------------------------------------------
// Kernel 1: fused QKV projection.  x[8192][1024] (f32) @ W (f32) -> bf16.
// blockIdx.z: 0=Q (scaled 1/8, [B,H,S,HD]), 1=K, 2=V transposed [B,H,HD,S].
// Software-pipelined: global loads for tile i+1 issue before compute of tile i;
// double-buffered LDS, one barrier per iteration.
// ---------------------------------------------------------------------------
__global__ __launch_bounds__(256, 1) void qkv_gemm(
    const float* __restrict__ x, const float* __restrict__ Wq,
    const float* __restrict__ Wk, const float* __restrict__ Wv,
    unsigned short* __restrict__ qo, unsigned short* __restrict__ ko,
    unsigned short* __restrict__ vto) {
  __shared__ __align__(16) unsigned short As[2][128][40];   // [m][k] bf16
  __shared__ __align__(16) unsigned short Ws[2][128][40];   // [n][k] bf16
  const int tid = threadIdx.x;
  const int m0 = blockIdx.x * 128;
  const int n0 = blockIdx.y * 128;
  const int mat = blockIdx.z;
  const float* W = (mat == 0) ? Wq : ((mat == 1) ? Wk : Wv);

  const int w = tid >> 5, lane = tid & 31;
  const int wm = w >> 1, wn = w & 1;
  const int h2 = lane >> 4, nl = lane & 15;

  // staging coordinates (fixed per thread)
  const int xr_r = tid >> 1, xr_c = (tid & 1) * 16;   // x: 128 rows x 32 cols, 16 f/thread
  const int wr_k = tid >> 3, wr_c = (tid & 7) * 16;   // W: 32 rows x 128 cols, 16 f/thread

  float4 xr[4], wr[4];
  auto glob_load = [&](int k0) {
    for (int p = 0; p < 4; ++p)
      xr[p] = *reinterpret_cast<const float4*>(x + (long)(m0 + xr_r) * DDIM + k0 + xr_c + p * 4);
    for (int p = 0; p < 4; ++p)
      wr[p] = *reinterpret_cast<const float4*>(W + (long)(k0 + wr_k) * DDIM + n0 + wr_c + p * 4);
  };
  auto lds_store = [&](int buf) {
    for (int p = 0; p < 4; ++p) {
      As[buf][xr_r][xr_c + p * 4 + 0] = f2bf(xr[p].x);
      As[buf][xr_r][xr_c + p * 4 + 1] = f2bf(xr[p].y);
      As[buf][xr_r][xr_c + p * 4 + 2] = f2bf(xr[p].z);
      As[buf][xr_r][xr_c + p * 4 + 3] = f2bf(xr[p].w);
    }
    for (int p = 0; p < 4; ++p) {
      Ws[buf][wr_c + p * 4 + 0][wr_k] = f2bf(wr[p].x);
      Ws[buf][wr_c + p * 4 + 1][wr_k] = f2bf(wr[p].y);
      Ws[buf][wr_c + p * 4 + 2][wr_k] = f2bf(wr[p].z);
      Ws[buf][wr_c + p * 4 + 3][wr_k] = f2bf(wr[p].w);
    }
  };

  f32x8 acc[2][4];
  for (int i = 0; i < 2; ++i)
    for (int j = 0; j < 4; ++j) acc[i][j] = zero8();

  glob_load(0);
  lds_store(0);
  __syncthreads();

  const int NIT = DDIM / 32;
  for (int it = 0; it < NIT; ++it) {
    const int cur = it & 1;
    if (it + 1 < NIT) {
      glob_load((it + 1) * 32);                  // issue next tile loads (no wait)
      if (it + 2 < NIT) {                        // prefetch tile i+2 (WGP scope)
        __builtin_prefetch(x + (long)(m0 + xr_r) * DDIM + (it + 2) * 32 + xr_c, 0, 3);
        __builtin_prefetch(W + (long)((it + 2) * 32 + wr_k) * DDIM + n0 + wr_c, 0, 3);
      }
    }
    // compute tile it: batch all fragment loads, then 8 WMMAs
    bf16x16 a0 = load_afrag(&As[cur][0][0], 40, wm * 32 + 0, 0);
    bf16x16 a1 = load_afrag(&As[cur][0][0], 40, wm * 32 + 16, 0);
    bf16x16 bfr[4];
    for (int j = 0; j < 4; ++j)
      bfr[j] = load_bfrag(&Ws[cur][0][0], 40, wn * 64 + j * 16, 0);
    for (int j = 0; j < 4; ++j) {
      acc[0][j] = wmma_bf16(a0, bfr[j], acc[0][j]);
      acc[1][j] = wmma_bf16(a1, bfr[j], acc[1][j]);
    }
    if (it + 1 < NIT) lds_store(cur ^ 1);        // loadcnt wait lands here
    __syncthreads();
  }

  // epilogue: one base pointer + stride per tile
  for (int i = 0; i < 2; ++i)
    for (int j = 0; j < 4; ++j) {
      int mg = m0 + wm * 32 + i * 16 + 8 * h2;
      int ng = n0 + wn * 64 + j * 16 + nl;
      int bidx = mg >> 11, sbase = mg & (SD - 1);
      int h = ng >> 6, hd = ng & (HDIM - 1);
      unsigned short* dst;
      long stride;
      float scl = 1.0f;
      if (mat == 2) {
        dst = vto + (((long)bidx * HH + h) * HDIM + hd) * SD + sbase;
        stride = 1;
      } else {
        unsigned short* base = (mat == 0) ? qo : ko;
        dst = base + (((long)bidx * HH + h) * SD + sbase) * HDIM + hd;
        stride = HDIM;
        if (mat == 0) scl = 0.125f;              // 1/sqrt(HD)
      }
      for (int r = 0; r < 8; ++r)
        dst[(long)r * stride] = f2bf(acc[i][j][r] * scl);
    }
}

// ---------------------------------------------------------------------------
// Kernel 2: causal flash attention with TDM-staged K/V tiles (double buffered).
// Block = 8 waves x 16 query rows; all waves sweep the block's key range.
// ---------------------------------------------------------------------------
__global__ __launch_bounds__(256) void flash_attn(
    const unsigned short* __restrict__ q, const unsigned short* __restrict__ k,
    const unsigned short* __restrict__ vt, unsigned short* __restrict__ ctx) {
  __shared__ __align__(16) unsigned short Ks[2][32][HDIM];  // keys x d
  __shared__ __align__(16) unsigned short Vs[2][HDIM][32];  // d x keys (from Vt)
  __shared__ __align__(16) unsigned short Ps[8][16][40];    // per-wave P relayout
  const int tid = threadIdx.x;
  const int w = tid >> 5, lane = tid & 31;
  const int nl = lane & 15, h2 = lane >> 4;
  const int hq = blockIdx.y, bq = blockIdx.z;
  const long bh = (long)bq * HH + hq;
  const unsigned short* qp = q + bh * SD * HDIM;
  const unsigned short* kp = k + bh * SD * HDIM;
  const unsigned short* vp = vt + bh * HDIM * SD;
  const int qbase = blockIdx.x * 128 + w * 16;
  const int qmax = qbase + 15;
  const int nsteps = (blockIdx.x * 128 + 128) >> 5;         // 32-key steps

  if (w == 0) {                                             // preload step 0 ASAP
    tdm_load_2d(&Ks[0][0][0], kp, HDIM, SD, HDIM, HDIM, 32);
    tdm_load_2d(&Vs[0][0][0], vp, SD, HDIM, SD, 32, HDIM);
  }

  bf16x16 aq0 = load_afrag(qp, HDIM, qbase, 0);             // overlaps with TDM
  bf16x16 aq1 = load_afrag(qp, HDIM, qbase, 32);

  f32x8 O[4];
  for (int t = 0; t < 4; ++t) O[t] = zero8();
  float mrow[8], lrow[8];
  for (int r = 0; r < 8; ++r) { mrow[r] = -1e30f; lrow[r] = 0.0f; }

  for (int it = 0; it < nsteps; ++it) {
    const int n0 = it << 5;
    const int cur = it & 1;
    if (w == 0) __builtin_amdgcn_s_wait_tensorcnt(0);       // step it's DMA done
    __syncthreads();                                        // publish buf[cur]
    if (w == 0 && it + 1 < nsteps) {                        // DMA next || compute
      const int nn = n0 + 32;
      tdm_load_2d(&Ks[cur ^ 1][0][0], kp + (long)nn * HDIM, HDIM, SD, HDIM, HDIM, 32);
      tdm_load_2d(&Vs[cur ^ 1][0][0], vp + nn, SD, HDIM, SD, 32, HDIM);
    }
    if (n0 <= qmax) {
      const unsigned short* Kb = &Ks[cur][0][0];
      const unsigned short* Vb = &Vs[cur][0][0];
      bf16x16 kb0 = load_bfrag(Kb, HDIM, 0, 0);             // batch all K frags
      bf16x16 kb1 = load_bfrag(Kb, HDIM, 0, 32);
      bf16x16 kb2 = load_bfrag(Kb, HDIM, 16, 0);
      bf16x16 kb3 = load_bfrag(Kb, HDIM, 16, 32);
      f32x8 s0 = zero8(), s1 = zero8();
      s0 = wmma_bf16(aq0, kb0, s0);
      s0 = wmma_bf16(aq1, kb1, s0);
      s1 = wmma_bf16(aq0, kb2, s1);
      s1 = wmma_bf16(aq1, kb3, s1);
      // V fragments issue now; ds latency hides under softmax VALU
      bf16x16 vb0 = load_bfrag(Vb, 32, 0, 0);
      bf16x16 vb1 = load_bfrag(Vb, 32, 16, 0);
      bf16x16 vb2 = load_bfrag(Vb, 32, 32, 0);
      bf16x16 vb3 = load_bfrag(Vb, 32, 48, 0);
      for (int r = 0; r < 8; ++r) {                         // online softmax
        const int qg = qbase + r + 8 * h2;
        float v0 = (n0 + nl      <= qg) ? s0[r] : -1e30f;   // causal mask
        float v1 = (n0 + 16 + nl <= qg) ? s1[r] : -1e30f;
        float tm = fmaxf(v0, v1);
        tm = fmaxf(tm, __shfl_xor(tm, 1, 32));
        tm = fmaxf(tm, __shfl_xor(tm, 2, 32));
        tm = fmaxf(tm, __shfl_xor(tm, 4, 32));
        tm = fmaxf(tm, __shfl_xor(tm, 8, 32));
        float mnew  = fmaxf(mrow[r], tm);
        float scale = __expf(mrow[r] - mnew);
        mrow[r] = mnew;
        float p0 = __expf(v0 - mnew);
        float p1 = __expf(v1 - mnew);
        float ps = p0 + p1;
        ps += __shfl_xor(ps, 1, 32);
        ps += __shfl_xor(ps, 2, 32);
        ps += __shfl_xor(ps, 4, 32);
        ps += __shfl_xor(ps, 8, 32);
        lrow[r] = lrow[r] * scale + ps;
        for (int t = 0; t < 4; ++t) O[t][r] *= scale;
        Ps[w][r + 8 * h2][nl]      = f2bf(p0);
        Ps[w][r + 8 * h2][16 + nl] = f2bf(p1);
      }
      bf16x16 ap = load_afrag(&Ps[w][0][0], 40, 0, 0);      // P as A-fragment
      O[0] = wmma_bf16(ap, vb0, O[0]);
      O[1] = wmma_bf16(ap, vb1, O[1]);
      O[2] = wmma_bf16(ap, vb2, O[2]);
      O[3] = wmma_bf16(ap, vb3, O[3]);
    }
    __syncthreads();                                        // done reading buf[cur]
  }

  for (int r = 0; r < 8; ++r) {                             // normalize + store
    float inv = 1.0f / lrow[r];
    int s = qbase + r + 8 * h2;
    unsigned short* dst = ctx + ((long)bq * SD + s) * DDIM + hq * HDIM + nl;
    for (int t = 0; t < 4; ++t)
      dst[t * 16] = f2bf(O[t][r] * inv);
  }
}

// ---------------------------------------------------------------------------
// Kernel 3: output projection. ctx(bf16) @ W_o(f32->bf16) + b_o -> f32 out.
// A tiles via TDM (with LDS padding to 40-ushort rows); W via register pipeline.
// ---------------------------------------------------------------------------
__global__ __launch_bounds__(256, 1) void out_proj(
    const unsigned short* __restrict__ ctx, const float* __restrict__ Wo,
    const float* __restrict__ bo, float* __restrict__ out) {
  __shared__ __align__(16) unsigned short As[2][128][40];
  __shared__ __align__(16) unsigned short Ws[2][128][40];
  const int tid = threadIdx.x;
  const int m0 = blockIdx.x * 128, n0 = blockIdx.y * 128;
  const int w = tid >> 5, lane = tid & 31;
  const int wm = w >> 1, wn = w & 1;
  const int h2 = lane >> 4, nl = lane & 15;

  const int wr_k = tid >> 3, wr_c = (tid & 7) * 16;

  float4 wr[4];
  auto wo_load = [&](int k0) {
    for (int p = 0; p < 4; ++p)
      wr[p] = *reinterpret_cast<const float4*>(Wo + (long)(k0 + wr_k) * DDIM + n0 + wr_c + p * 4);
  };
  auto wo_store = [&](int buf) {
    for (int p = 0; p < 4; ++p) {
      Ws[buf][wr_c + p * 4 + 0][wr_k] = f2bf(wr[p].x);
      Ws[buf][wr_c + p * 4 + 1][wr_k] = f2bf(wr[p].y);
      Ws[buf][wr_c + p * 4 + 2][wr_k] = f2bf(wr[p].z);
      Ws[buf][wr_c + p * 4 + 3][wr_k] = f2bf(wr[p].w);
    }
  };

  f32x8 acc[2][4];
  for (int i = 0; i < 2; ++i)
    for (int j = 0; j < 4; ++j) acc[i][j] = zero8();

  // preload tile 0: A via TDM (padded rows), W via registers
  if (w == 0)
    tdm_load_2d(&As[0][0][0], ctx + (long)m0 * DDIM, DDIM, BD * SD, DDIM, 32, 128,
                TDM_PAD_40);
  wo_load(0);
  wo_store(0);

  const int NIT = DDIM / 32;
  for (int it = 0; it < NIT; ++it) {
    const int cur = it & 1;
    if (w == 0) __builtin_amdgcn_s_wait_tensorcnt(0);       // A tile it ready
    __syncthreads();                                        // publish both tiles
    if (it + 1 < NIT) {
      if (w == 0)
        tdm_load_2d(&As[cur ^ 1][0][0], ctx + (long)m0 * DDIM + (it + 1) * 32,
                    DDIM, BD * SD, DDIM, 32, 128, TDM_PAD_40);
      wo_load((it + 1) * 32);                               // issue, no wait
      if (it + 2 < NIT)
        __builtin_prefetch(Wo + (long)((it + 2) * 32 + wr_k) * DDIM + n0 + wr_c, 0, 3);
    }
    bf16x16 a0 = load_afrag(&As[cur][0][0], 40, wm * 32 + 0, 0);
    bf16x16 a1 = load_afrag(&As[cur][0][0], 40, wm * 32 + 16, 0);
    bf16x16 bfr[4];
    for (int j = 0; j < 4; ++j)
      bfr[j] = load_bfrag(&Ws[cur][0][0], 40, wn * 64 + j * 16, 0);
    for (int j = 0; j < 4; ++j) {
      acc[0][j] = wmma_bf16(a0, bfr[j], acc[0][j]);
      acc[1][j] = wmma_bf16(a1, bfr[j], acc[1][j]);
    }
    if (it + 1 < NIT) wo_store(cur ^ 1);                    // loadcnt wait here
    __syncthreads();
  }

  for (int i = 0; i < 2; ++i)
    for (int j = 0; j < 4; ++j) {
      int mg = m0 + wm * 32 + i * 16 + 8 * h2;
      int ng = n0 + wn * 64 + j * 16 + nl;
      float* dst = out + (long)mg * DDIM + ng;
      float bias = bo[ng];
      for (int r = 0; r < 8; ++r)
        dst[(long)r * DDIM] = acc[i][j][r] + bias;
    }
}

// ---------------------------------------------------------------------------
extern "C" void kernel_launch(void* const* d_in, const int* in_sizes, int n_in,
                              void* d_out, int out_size, void* d_ws, size_t ws_size,
                              hipStream_t stream) {
  const float* x  = (const float*)d_in[0];
  const float* Wq = (const float*)d_in[1];
  const float* Wk = (const float*)d_in[2];
  const float* Wv = (const float*)d_in[3];
  const float* Wo = (const float*)d_in[4];
  const float* bo = (const float*)d_in[5];

  // Workspace: 4 bf16 buffers of B*H*S*HD = 8Mi elements each (64 MB total).
  const long NE = (long)BD * HH * SD * HDIM;
  unsigned short* qw  = (unsigned short*)d_ws;
  unsigned short* kw  = qw + NE;
  unsigned short* vtw = kw + NE;
  unsigned short* cw  = vtw + NE;
  float* out = (float*)d_out;

  qkv_gemm<<<dim3((BD * SD) / 128, DDIM / 128, 3), 256, 0, stream>>>(x, Wq, Wk, Wv, qw, kw, vtw);
  flash_attn<<<dim3(SD / 128, HH, BD), 256, 0, stream>>>(qw, kw, vtw, cw);
  out_proj<<<dim3((BD * SD) / 128, DDIM / 128), 256, 0, stream>>>(cw, Wo, bo, out);
}
